// ScaledDotProduct_61512521614064
// MI455X (gfx1250) — matrix operand order repaired and checked by reading, not compiled, not run on hardware
//
#include <hip/hip_runtime.h>
#include <hip/hip_bf16.h>

// ---- types -----------------------------------------------------------------
typedef __attribute__((ext_vector_type(16))) __bf16        v16bf;
typedef __attribute__((ext_vector_type(8)))  float         v8f;
typedef __attribute__((ext_vector_type(8)))  unsigned short u16x8;

#define WAVES    4
#define BLOCK_M  128      // 4 waves x 32 query rows
#define KT       32       // keys per streamed tile
#define Dh       64
#define S_LEN    2048

union FragB {
    v16bf v;
    u16x8 h[2];
    unsigned short u[16];
};

__device__ __forceinline__ unsigned short f2bf(float f) {
    unsigned int u = __float_as_uint(f);
    u += 0x7FFFu + ((u >> 16) & 1u);     // round-to-nearest-even
    return (unsigned short)(u >> 16);
}

// ---- flash attention forward (causal, ghostmax) ----------------------------
__global__ void __launch_bounds__(128, 1)
fa_fwd_ghostmax(const float* __restrict__ Q, const float* __restrict__ K,
                const float* __restrict__ V, float* __restrict__ O)
{
    __shared__ alignas(64) unsigned short Kb[2][KT * Dh];     // [key][d]   bf16
    __shared__ alignas(64) unsigned short Vb[2][Dh * KT];     // [d][key]   bf16 (transposed)
    __shared__ alignas(64) unsigned short Pb[WAVES][16 * KT]; // per-wave P bounce

    const int tid  = threadIdx.x;
    const int wid  = tid >> 5;
    const int lane = tid & 31;
    const int half = lane >> 4;     // 0: lanes 0-15, 1: lanes 16-31
    const int lcol = lane & 15;

    const int    bh      = blockIdx.y;
    const int    m_start = blockIdx.x * BLOCK_M;
    const size_t base    = (size_t)bh * S_LEN * Dh;

    const float SCALE = 0.125f * 1.4426950408889634f;   // (1/sqrt(64)) * log2(e)
    const float NEGI  = -1.0e9f * 1.4426950408889634f;  // additive mask in log2 domain

    // ---- Q fragments, kept resident in VGPRs --------------------------------
    // A layout (16x32 bf16): M = lane&15; element e K = half*16 + e
    FragB qa[2][2];
    #pragma unroll
    for (int mt = 0; mt < 2; ++mt) {
        const int    qrow = m_start + wid * 32 + mt * 16 + lcol;
        const float* qp   = Q + base + (size_t)qrow * Dh;
        #pragma unroll
        for (int s = 0; s < 2; ++s) {
            const int d0 = s * 32 + half * 16;
            const int d1 = s * 32 + half * 16 + 8;
            const float4* p0 = (const float4*)(qp + d0);
            const float4* p1 = (const float4*)(qp + d1);
            float4 a = p0[0], b = p0[1], c = p1[0], d = p1[1];
            const float lo[8] = {a.x,a.y,a.z,a.w,b.x,b.y,b.z,b.w};
            const float hi[8] = {c.x,c.y,c.z,c.w,d.x,d.y,d.z,d.w};
            #pragma unroll
            for (int e = 0; e < 8; ++e) {
                qa[mt][s].u[e]     = f2bf(lo[e]);
                qa[mt][s].u[8 + e] = f2bf(hi[e]);
            }
        }
    }

    // all-ones B fragment: row-sum of P via the matrix pipe (rs = P x 1)
    FragB ones;
    #pragma unroll
    for (int e = 0; e < 16; ++e) ones.u[e] = 0x3F80;   // bf16 1.0

    // ---- accumulators + online-softmax state (ghost logit: m=0, l=1) -------
    v8f   acc[2][4];
    float m_run[2][8], l_run[2][8];
    #pragma unroll
    for (int mt = 0; mt < 2; ++mt) {
        #pragma unroll
        for (int nt = 0; nt < 4; ++nt) acc[mt][nt] = (v8f){0,0,0,0,0,0,0,0};
        #pragma unroll
        for (int r = 0; r < 8; ++r) { m_run[mt][r] = 0.0f; l_run[mt][r] = 1.0f; }
    }

    // ---- K/V staging assignment (coalesced b128 loads) ----------------------
    const int    skey = tid >> 2;          // 0..31
    const int    sdq  = (tid & 3) * 16;    // 0,16,32,48
    const float* Kg   = K + base;
    const float* Vg   = V + base;

    float4 kreg[4], vreg[4];
    auto stage = [&](int jt) {
        const size_t  off = (size_t)(jt * KT + skey) * Dh + sdq;
        const float4* kp  = (const float4*)(Kg + off);
        const float4* vp  = (const float4*)(Vg + off);
        #pragma unroll
        for (int i = 0; i < 4; ++i) { kreg[i] = kp[i]; vreg[i] = vp[i]; }
    };
    auto commit = [&](int buf) {
        const float* kf = (const float*)kreg;
        const float* vf = (const float*)vreg;
        alignas(16) unsigned short tmp[16];
        #pragma unroll
        for (int e = 0; e < 16; ++e) tmp[e] = f2bf(kf[e]);
        *(u16x8*)&Kb[buf][skey * Dh + sdq]     = *(const u16x8*)&tmp[0];
        *(u16x8*)&Kb[buf][skey * Dh + sdq + 8] = *(const u16x8*)&tmp[8];
        #pragma unroll
        for (int e = 0; e < 16; ++e)           // transpose into [d][key]
            Vb[buf][(sdq + e) * KT + skey] = f2bf(vf[e]);
    };

    const int ntiles = m_start / KT + BLOCK_M / KT;   // causal: stop at diagonal

    stage(0);
    commit(0);

    for (int jt = 0; jt < ntiles; ++jt) {
        const int buf = jt & 1;
        const int j0  = jt * KT;
        __syncthreads();                       // buf[jt&1] ready for everyone

        const bool more = (jt + 1 < ntiles);
        if (more) stage(jt + 1);               // HBM latency hidden under WMMA
        if (jt + 2 < ntiles) {                 // L2 prefetch of tile jt+2
            const size_t poff = (size_t)((jt + 2) * KT + skey) * Dh + sdq;
            __builtin_prefetch(Kg + poff, 0, 1);
            __builtin_prefetch(Vg + poff, 0, 1);
        }
        const bool needmask = (j0 + KT - 1 > m_start);

        // B fragments of K^T: element e <-> d = s*32 + half*16 + e ; col = key
        FragB kf[2][2];
        #pragma unroll
        for (int nt = 0; nt < 2; ++nt)
            #pragma unroll
            for (int s = 0; s < 2; ++s)
                kf[nt][s].v = *(const v16bf*)&Kb[buf][(nt * 16 + lcol) * Dh + s * 32 + half * 16];

        #pragma unroll
        for (int mt = 0; mt < 2; ++mt) {
            // ---- S = Q K^T (two 16-key sub-tiles, K-dim = 64 = 2 wmma each)
            v8f s0 = (v8f){0,0,0,0,0,0,0,0};
            v8f s1 = (v8f){0,0,0,0,0,0,0,0};
            s0 = __builtin_amdgcn_wmma_f32_16x16x32_bf16(false, qa[mt][0].v, false, kf[0][0].v, (short)0, s0, false, false);
            s0 = __builtin_amdgcn_wmma_f32_16x16x32_bf16(false, qa[mt][1].v, false, kf[0][1].v, (short)0, s0, false, false);
            s1 = __builtin_amdgcn_wmma_f32_16x16x32_bf16(false, qa[mt][0].v, false, kf[1][0].v, (short)0, s1, false, false);
            s1 = __builtin_amdgcn_wmma_f32_16x16x32_bf16(false, qa[mt][1].v, false, kf[1][1].v, (short)0, s1, false, false);

            // ---- scale + causal mask (C layout: row = r + 8*half, col = lcol)
            float a0[8], a1[8];
            #pragma unroll
            for (int r = 0; r < 8; ++r) { a0[r] = s0[r] * SCALE; a1[r] = s1[r] * SCALE; }
            if (needmask) {
                const int rowbase = m_start + wid * 32 + mt * 16 + half * 8;
                const int k0 = j0 + lcol, k1 = j0 + 16 + lcol;
                #pragma unroll
                for (int r = 0; r < 8; ++r) {
                    const int row = rowbase + r;
                    if (k0 > row) a0[r] += NEGI;
                    if (k1 > row) a1[r] += NEGI;
                }
            }

            // ---- online ghost-softmax: row max via half-wave shuffles
            float rm[8];
            #pragma unroll
            for (int r = 0; r < 8; ++r) rm[r] = fmaxf(a0[r], a1[r]);
            #pragma unroll
            for (int off = 1; off < 16; off <<= 1)
                #pragma unroll
                for (int r = 0; r < 8; ++r)
                    rm[r] = fmaxf(rm[r], __shfl_xor(rm[r], off, 32));

            float alpha[8], p0[8], p1[8];
            #pragma unroll
            for (int r = 0; r < 8; ++r) {
                const float mn = fmaxf(m_run[mt][r], rm[r]);
                alpha[r]     = exp2f(m_run[mt][r] - mn);
                m_run[mt][r] = mn;
                p0[r] = exp2f(a0[r] - mn);
                p1[r] = exp2f(a1[r] - mn);
            }
            #pragma unroll
            for (int nt = 0; nt < 4; ++nt)
                #pragma unroll
                for (int r = 0; r < 8; ++r) acc[mt][nt][r] *= alpha[r];

            // ---- C-layout -> A-layout for P via per-wave LDS bounce
            #pragma unroll
            for (int r = 0; r < 8; ++r) {
                Pb[wid][(r + 8 * half) * KT + lcol]      = f2bf(p0[r]);
                Pb[wid][(r + 8 * half) * KT + 16 + lcol] = f2bf(p1[r]);
            }
            asm volatile("s_wait_dscnt 0x0" ::: "memory");   // wave-local RAW on LDS
            FragB pa;
            pa.h[0] = *(const u16x8*)&Pb[wid][lcol * KT + half * 16];
            pa.h[1] = *(const u16x8*)&Pb[wid][lcol * KT + half * 16 + 8];

            // ---- row sums on the matrix pipe: rs = P x ones (replicated per row)
            v8f rs = (v8f){0,0,0,0,0,0,0,0};
            rs = __builtin_amdgcn_wmma_f32_16x16x32_bf16(false, pa.v, false, ones.v, (short)0, rs, false, false);
            #pragma unroll
            for (int r = 0; r < 8; ++r)
                l_run[mt][r] = l_run[mt][r] * alpha[r] + rs[r];

            // ---- O += P V  (B frag: element e <-> key = half*16 + e ; col = d)
            #pragma unroll
            for (int nt = 0; nt < 4; ++nt) {
                FragB vf;
                vf.v = *(const v16bf*)&Vb[buf][(nt * 16 + lcol) * KT + half * 16];
                acc[mt][nt] = __builtin_amdgcn_wmma_f32_16x16x32_bf16(
                    false, pa.v, false, vf.v, (short)0, acc[mt][nt], false, false);
            }
        }

        if (more) commit((jt + 1) & 1);   // double buffer: no second barrier needed
    }

    // ---- epilogue: divide by ghost-softmax denominator, write out -----------
    #pragma unroll
    for (int mt = 0; mt < 2; ++mt) {
        float inv[8];
        #pragma unroll
        for (int r = 0; r < 8; ++r) inv[r] = 1.0f / l_run[mt][r];
        const int rowbase = m_start + wid * 32 + mt * 16 + half * 8;
        #pragma unroll
        for (int nt = 0; nt < 4; ++nt)
            #pragma unroll
            for (int r = 0; r < 8; ++r)
                O[base + (size_t)(rowbase + r) * Dh + nt * 16 + lcol] = acc[mt][nt][r] * inv[r];
    }
}

// ---- launch ----------------------------------------------------------------
extern "C" void kernel_launch(void* const* d_in, const int* in_sizes, int n_in,
                              void* d_out, int out_size, void* d_ws, size_t ws_size,
                              hipStream_t stream) {
    (void)in_sizes; (void)n_in; (void)out_size; (void)d_ws; (void)ws_size;
    const float* q = (const float*)d_in[0];
    const float* k = (const float*)d_in[1];
    const float* v = (const float*)d_in[2];
    float*       o = (float*)d_out;

    const int B = 2, H = 16;
    dim3 grid(S_LEN / BLOCK_M, B * H);   // (16, 32)
    dim3 block(128);
    fa_fwd_ghostmax<<<grid, block, 0, stream>>>(q, k, v, o);
}
